// GRU_ODE_75531294867998
// MI455X (gfx1250) — matrix operand order repaired and checked by reading, not compiled, not run on hardware
//
#include <hip/hip_runtime.h>
#include <hip/hip_bf16.h>

// ---------- types ----------
typedef __attribute__((ext_vector_type(16))) __bf16 v16bf;
typedef __attribute__((ext_vector_type(8)))  float  v8f;
typedef __attribute__((ext_vector_type(4)))  unsigned int v4u;

union ABFrag {
    v4u            q[2];   // 32 bytes
    unsigned short u[16];
    v16bf          v;
};

__device__ __forceinline__ unsigned short f2bf(float f) {
    unsigned int x = __float_as_uint(f);
    x += 0x7FFFu + ((x >> 16) & 1u);           // round-to-nearest-even
    return (unsigned short)(x >> 16);
}
__device__ __forceinline__ float sigmoidf_(float x) {
    return __builtin_amdgcn_rcpf(1.0f + __expf(-x));      // v_exp_f32 + v_rcp_f32
}
__device__ __forceinline__ float tanhf_(float x) {
    return 2.0f * sigmoidf_(2.0f * x) - 1.0f;
}

__device__ __forceinline__ void vzero4x8(v8f acc[4]) {
#pragma unroll
    for (int n = 0; n < 4; ++n)
#pragma unroll
        for (int i = 0; i < 8; ++i) acc[n][i] = 0.0f;
}

__device__ __forceinline__ v8f wmma_bf16(const ABFrag& a, const ABFrag& b, v8f c) {
    return __builtin_amdgcn_wmma_f32_16x16x32_bf16(
        /*neg_a=*/false, a.v, /*neg_b=*/false, b.v,
        /*c_mod=*/(short)0, c, /*reuse_a=*/false, /*reuse_b=*/false);
}

// A fragment (16x32 MxK) from a bf16 LDS tile.
// ISA layout: lane<16 -> M=lane, K = kb+{0..7, 16..23}; lane>=16 -> M=lane-16, K = kb+{8..15, 24..31}
template <int STRIDE>
__device__ __forceinline__ ABFrag load_a_bf16(const unsigned short* base, int mr, int kb, int lane) {
    ABFrag f;
    const int m  = mr + (lane & 15);
    const int h0 = (lane >> 4) << 3;
    const unsigned short* p = base + m * STRIDE + kb + h0;
    f.q[0] = *(const v4u*)(p);
    f.q[1] = *(const v4u*)(p + 16);
    return f;
}

// B fragment (32x16 KxN) from pre-swizzled packed weights:
// tile (kt,nt) stored as 32 lanes x 16 contiguous bf16 (lane = n%16 + 16*((kr/8)%2), elem i = kr%8 + 8*(kr>=16)).
__device__ __forceinline__ ABFrag load_b(const unsigned short* Wpk, int ntiles_n, int kt, int nt, int lane) {
    const unsigned short* p = Wpk + ((size_t)(kt * ntiles_n + nt) * 32 + lane) * 16;
    ABFrag f;
    f.q[0] = *(const v4u*)(p);
    f.q[1] = *(const v4u*)(p + 8);
    return f;
}

// ---- sequential single-accumulator GEMM pass (low VGPR pressure) ----
template <int STRIDE>
__device__ __forceinline__ void gemm_bf16(v8f acc[4], const unsigned short* abase, int mr,
                                          const unsigned short* Wpk, int cq, int lane) {
#pragma unroll 1
    for (int kt = 0; kt < 8; ++kt) {
        ABFrag a = load_a_bf16<STRIDE>(abase, mr, kt * 32, lane);
#pragma unroll
        for (int n = 0; n < 4; ++n)
            acc[n] = wmma_bf16(a, load_b(Wpk, 16, kt, cq * 4 + n, lane), acc[n]);
    }
}
// single K=32 step from the (32-wide) x_in tile
__device__ __forceinline__ void gemm_x(v8f acc[4], const unsigned short* xbase, int mr,
                                       const unsigned short* Wpk, int cq, int lane) {
    ABFrag a = load_a_bf16<32>(xbase, mr, 0, lane);
#pragma unroll
    for (int n = 0; n < 4; ++n)
        acc[n] = wmma_bf16(a, load_b(Wpk, 16, 0, cq * 4 + n, lane), acc[n]);
}

// ---------- pack: f32 (K,N) row-major -> bf16 swizzled tiles ----------
__global__ void pack_w_kernel(const float* __restrict__ W, unsigned short* __restrict__ dst,
                              int K, int N, int Npad) {
    int idx = blockIdx.x * 256 + threadIdx.x;
    int total = K * Npad;
    if (idx >= total) return;
    int k = idx / Npad, n = idx - k * Npad;
    float v = (n < N) ? W[(size_t)k * N + n] : 0.0f;
    int kt = k >> 5, kr = k & 31;
    int lane = (n & 15) + (((kr >> 3) & 1) << 4);
    int i = (kr & 7) + ((kr >> 4) << 3);
    int nt = n >> 4;
    dst[((size_t)(kt * (Npad >> 4) + nt) * 32 + lane) * 16 + i] = f2bf(v);
}

__global__ void init_kernel(float* __restrict__ h, float* __restrict__ acc) {
    int idx = blockIdx.x * 256 + threadIdx.x;
    if (idx < 1024 * 256) h[idx] = 0.0f;
    if (idx < 4) acc[idx] = 0.0f;
}

// ---------- kernel A: 4 fused Euler steps for a 32-row tile ----------
// r = sig(h@Wr+br); z = sig(h@Wz+bz); u = tanh((r*h)@Wh+bh); h += dt*(1-z)*(u-h)
// f32 state lives in per-lane registers (each lane owns its 32 C-fragment elements);
// a bf16 shadow of the tile lives in LDS and feeds the WMMA A operand.
__global__ __launch_bounds__(256)
__attribute__((amdgpu_waves_per_eu(4)))
void euler4_kernel(
    float* __restrict__ h,
    const unsigned short* __restrict__ Wr, const unsigned short* __restrict__ Wz,
    const unsigned short* __restrict__ Wh,
    const float* __restrict__ br, const float* __restrict__ bz, const float* __restrict__ bh,
    const float* __restrict__ dtp) {
    __shared__ unsigned short hbf[32 * 272];
    __shared__ unsigned short rhbuf[32 * 272];

    const int t = threadIdx.x, lane = t & 31, w = t >> 5;
    const int rb = w >> 2, cq = w & 3;     // 2 row-blocks x 4 col-quarters
    const int m0 = blockIdx.x * 32;
    const int mr = rb * 16;
    const int colb = cq * 64 + (lane & 15);
    const int rowb = mr + ((lane >> 4) << 3);
    const float dt = dtp[0];

    // coalesced staging of the bf16 shadow
    for (int jj = 0; jj < 32; ++jj)
        hbf[jj * 272 + t] = f2bf(h[(size_t)(m0 + jj) * 256 + t]);
    // exact f32 state for this lane's owned elements
    float hreg[4][8];
#pragma unroll
    for (int n = 0; n < 4; ++n)
#pragma unroll
        for (int ci = 0; ci < 8; ++ci)
            hreg[n][ci] = h[(size_t)(m0 + rowb + ci) * 256 + colb + n * 16];
    __syncthreads();

#pragma unroll 1
    for (int step = 0; step < 4; ++step) {
        v8f acc[4];
        // ---- pass r: gr = h@Wr; write rh = r*h (exact f32 h) to LDS, drop r ----
        vzero4x8(acc);
        gemm_bf16<272>(acc, hbf, mr, Wr, cq, lane);
#pragma unroll
        for (int n = 0; n < 4; ++n) {
            const int col = colb + n * 16;
            const float brv = br[col];
#pragma unroll
            for (int ci = 0; ci < 8; ++ci) {
                float r = sigmoidf_(acc[n][ci] + brv);
                rhbuf[(rowb + ci) * 272 + col] = f2bf(r * hreg[n][ci]);
            }
        }
        // ---- pass z: gz = h@Wz; keep z in regs ----
        vzero4x8(acc);
        gemm_bf16<272>(acc, hbf, mr, Wz, cq, lane);
        float zreg[4][8];
#pragma unroll
        for (int n = 0; n < 4; ++n) {
            const float bzv = bz[colb + n * 16];
#pragma unroll
            for (int ci = 0; ci < 8; ++ci) zreg[n][ci] = sigmoidf_(acc[n][ci] + bzv);
        }
        __syncthreads();   // rh visible to all; all reads of hbf for this step done
        // ---- pass u: u = (r*h)@Wh ; h += dt*(1-z)*(u-h) ; refresh bf16 shadow ----
        vzero4x8(acc);
        gemm_bf16<272>(acc, rhbuf, mr, Wh, cq, lane);
#pragma unroll
        for (int n = 0; n < 4; ++n) {
            const int col = colb + n * 16;
            const float bhv = bh[col];
#pragma unroll
            for (int ci = 0; ci < 8; ++ci) {
                float u = tanhf_(acc[n][ci] + bhv);
                float hv = hreg[n][ci];
                hv += dt * (1.0f - zreg[n][ci]) * (u - hv);
                hreg[n][ci] = hv;
                hbf[(rowb + ci) * 272 + col] = f2bf(hv);
            }
        }
        __syncthreads();
    }
#pragma unroll
    for (int n = 0; n < 4; ++n)
#pragma unroll
        for (int ci = 0; ci < 8; ++ci)
            h[(size_t)(m0 + rowb + ci) * 256 + colb + n * 16] = hreg[n][ci];
}

// ---------- kernel B: per-event gather + prediction/loss + GRU update + scatter ----------
__global__ __launch_bounds__(256)
__attribute__((amdgpu_waves_per_eu(4)))
void event_kernel(
    float* __restrict__ h,
    const float* __restrict__ X, const float* __restrict__ M,
    const int* __restrict__ bidx, int e,
    const unsigned short* __restrict__ Uhr, const unsigned short* __restrict__ Uhz,
    const unsigned short* __restrict__ Uhn,
    const unsigned short* __restrict__ Wxr, const unsigned short* __restrict__ Wxz,
    const unsigned short* __restrict__ Wxn,
    const unsigned short* __restrict__ W1, const unsigned short* __restrict__ W2,
    const float* __restrict__ bxr, const float* __restrict__ bxz, const float* __restrict__ bxn,
    const float* __restrict__ bhr, const float* __restrict__ bhz, const float* __restrict__ bhn,
    const float* __restrict__ pb1, const float* __restrict__ pb2,
    float* __restrict__ acc_g) {
    __shared__ unsigned short hibf[32 * 272];
    __shared__ unsigned short actbuf[32 * 272];
    __shared__ unsigned short xbuf[32 * 32];
    __shared__ int            iebuf[32];
    __shared__ float          accbuf[4];

    const int t = threadIdx.x, lane = t & 31, w = t >> 5;
    const int rb = w >> 2, cq = w & 3;
    const int j0 = blockIdx.x * 32;
    const int mr = rb * 16;
    const int colb = cq * 64 + (lane & 15);
    const int rowb = mr + ((lane >> 4) << 3);

    if (t < 32) iebuf[t] = bidx[e * 512 + j0 + t];
    if (t < 4) accbuf[t] = 0.0f;
    {   // x_in = Xe.reshape(NO, 32) -> bf16 LDS
        const float* Xe = X + (size_t)(e * 512 + j0) * 32;
#pragma unroll
        for (int q = 0; q < 4; ++q) {
            int idx = t * 4 + q;
            xbuf[idx] = f2bf(Xe[idx]);
        }
    }
    __syncthreads();
    // gather: coalesced bf16 staging + exact f32 copies of owned elements
    for (int jj = 0; jj < 32; ++jj)
        hibf[jj * 272 + t] = f2bf(h[(size_t)iebuf[jj] * 256 + t]);
    float hireg[4][8];
#pragma unroll
    for (int n = 0; n < 4; ++n)
#pragma unroll
        for (int ci = 0; ci < 8; ++ci)
            hireg[n][ci] = h[(size_t)iebuf[rowb + ci] * 256 + colb + n * 16];
    __syncthreads();

    v8f acc[4];
    // ---- prediction: act = relu(hi@W1 + b1) ----
    vzero4x8(acc);
    gemm_bf16<272>(acc, hibf, mr, W1, cq, lane);
#pragma unroll
    for (int n = 0; n < 4; ++n) {
        const int col = colb + n * 16;
        const float b = pb1[col];
#pragma unroll
        for (int ci = 0; ci < 8; ++ci) {
            float v = acc[n][ci] + b;
            actbuf[(rowb + ci) * 272 + col] = f2bf(v > 0.0f ? v : 0.0f);
        }
    }
    __syncthreads();

    // ---- p = act@W2 + b2 ; loss terms (cols 0..14 valid) ----
    float lmse = 0.0f, lmae = 0.0f, lmape = 0.0f, ltot = 0.0f;
    if (w < 2) {
        v8f p;
#pragma unroll
        for (int i = 0; i < 8; ++i) p[i] = 0.0f;
        const int mr2 = w * 16;
#pragma unroll 1
        for (int kt = 0; kt < 8; ++kt) {
            ABFrag a = load_a_bf16<272>(actbuf, mr2, kt * 32, lane);
            p = wmma_bf16(a, load_b(W2, 1, kt, 0, lane), p);
        }
        const int n = lane & 15;
        if (n < 15) {
            const float b2 = pb2[n];
#pragma unroll
            for (int ci = 0; ci < 8; ++ci) {
                const int row = mr2 + ci + ((lane >> 4) << 3);
                const int j = j0 + row;
                float pe = p[ci] + b2;
                float Xt = X[(((size_t)e * 512 + j) * 16 + (n + 1)) * 2];
                float Mt = M[((size_t)e * 512 + j) * 16 + (n + 1)];
                float d = Xt - pe;
                lmse += d * d * Mt;
                lmae += fabsf(d) * Mt;
                lmape += fabsf(d) / (Xt + 1e-8f) * Mt;
                ltot += Mt;
            }
        }
    }
    atomicAdd(&accbuf[0], lmse);
    atomicAdd(&accbuf[1], lmae);
    atomicAdd(&accbuf[2], lmape);
    atomicAdd(&accbuf[3], ltot);

    // ---- GRU pass r: r = sig(x@Wxr + hi@Uhr + bxr + bhr) ----
    float rreg[4][8], zreg[4][8];
    vzero4x8(acc);
    gemm_x(acc, xbuf, mr, Wxr, cq, lane);
    gemm_bf16<272>(acc, hibf, mr, Uhr, cq, lane);
#pragma unroll
    for (int n = 0; n < 4; ++n) {
        const float b_r = bxr[colb + n * 16] + bhr[colb + n * 16];
#pragma unroll
        for (int ci = 0; ci < 8; ++ci) rreg[n][ci] = sigmoidf_(acc[n][ci] + b_r);
    }
    // ---- GRU pass z ----
    vzero4x8(acc);
    gemm_x(acc, xbuf, mr, Wxz, cq, lane);
    gemm_bf16<272>(acc, hibf, mr, Uhz, cq, lane);
#pragma unroll
    for (int n = 0; n < 4; ++n) {
        const float b_z = bxz[colb + n * 16] + bhz[colb + n * 16];
#pragma unroll
        for (int ci = 0; ci < 8; ++ci) zreg[n][ci] = sigmoidf_(acc[n][ci] + b_z);
    }
    // ---- GRU pass n: n = tanh(x@Wxn + bxn + r*(hi@Uhn + bhn)); hn = (1-z)*n + z*hi ----
    vzero4x8(acc);
    gemm_bf16<272>(acc, hibf, mr, Uhn, cq, lane);
#pragma unroll
    for (int n = 0; n < 4; ++n) {
        const float bhnv = bhn[colb + n * 16];
#pragma unroll
        for (int ci = 0; ci < 8; ++ci) acc[n][ci] = rreg[n][ci] * (acc[n][ci] + bhnv);
    }
    gemm_x(acc, xbuf, mr, Wxn, cq, lane);
    // scatter hn straight to global h (disjoint ownership; no one reads h after gather)
#pragma unroll
    for (int n = 0; n < 4; ++n) {
        const int col = colb + n * 16;
        const float bxnv = bxn[col];
#pragma unroll
        for (int ci = 0; ci < 8; ++ci) {
            float nn = tanhf_(acc[n][ci] + bxnv);
            float hn = (1.0f - zreg[n][ci]) * nn + zreg[n][ci] * hireg[n][ci];
            h[(size_t)iebuf[rowb + ci] * 256 + col] = hn;
        }
    }
    __syncthreads();
    if (t < 4) atomicAdd(&acc_g[t], accbuf[t]);
}

__global__ void final_kernel(const float* __restrict__ acc, float* __restrict__ out) {
    if (threadIdx.x == 0) {
        float tot = acc[3];
        out[0] = acc[0] / tot;
        out[1] = acc[1] / tot;
        out[2] = acc[2] / tot;
    }
}

// ---------- host ----------
extern "C" void kernel_launch(void* const* d_in, const int* in_sizes, int n_in,
                              void* d_out, int out_size, void* d_ws, size_t ws_size,
                              hipStream_t stream) {
    (void)in_sizes; (void)n_in; (void)out_size; (void)ws_size;
    const float* X      = (const float*)d_in[0];
    const float* M      = (const float*)d_in[1];
    const int*   bidx   = (const int*)d_in[2];
    const float* dt     = (const float*)d_in[3];
    const float* ode_Wr = (const float*)d_in[4];
    const float* ode_br = (const float*)d_in[5];
    const float* ode_Wz = (const float*)d_in[6];
    const float* ode_bz = (const float*)d_in[7];
    const float* ode_Wh = (const float*)d_in[8];
    const float* ode_bh = (const float*)d_in[9];
    const float* g_Wxr  = (const float*)d_in[10];
    const float* g_Wxz  = (const float*)d_in[11];
    const float* g_Wxn  = (const float*)d_in[12];
    const float* g_Uhr  = (const float*)d_in[13];
    const float* g_Uhz  = (const float*)d_in[14];
    const float* g_Uhn  = (const float*)d_in[15];
    const float* g_bxr  = (const float*)d_in[16];
    const float* g_bxz  = (const float*)d_in[17];
    const float* g_bxn  = (const float*)d_in[18];
    const float* g_bhr  = (const float*)d_in[19];
    const float* g_bhz  = (const float*)d_in[20];
    const float* g_bhn  = (const float*)d_in[21];
    const float* p_W1   = (const float*)d_in[22];
    const float* p_b1   = (const float*)d_in[23];
    const float* p_W2   = (const float*)d_in[24];
    const float* p_b2   = (const float*)d_in[25];

    char*  ws  = (char*)d_ws;
    float* h   = (float*)ws;                       // 1024*256*4 = 1 MB
    float* acc = (float*)(ws + (1u << 20));        // 4 floats
    size_t off = (1u << 20) + 256;
    auto take = [&](size_t bytes) -> unsigned short* {
        unsigned short* p = (unsigned short*)(ws + off);
        off += (bytes + 255) & ~(size_t)255;
        return p;
    };
    unsigned short* Wr_pk  = take(256 * 256 * 2);
    unsigned short* Wz_pk  = take(256 * 256 * 2);
    unsigned short* Wh_pk  = take(256 * 256 * 2);
    unsigned short* Uhr_pk = take(256 * 256 * 2);
    unsigned short* Uhz_pk = take(256 * 256 * 2);
    unsigned short* Uhn_pk = take(256 * 256 * 2);
    unsigned short* W1_pk  = take(256 * 256 * 2);
    unsigned short* Wxr_pk = take(32 * 256 * 2);
    unsigned short* Wxz_pk = take(32 * 256 * 2);
    unsigned short* Wxn_pk = take(32 * 256 * 2);
    unsigned short* W2_pk  = take(256 * 16 * 2);

    hipLaunchKernelGGL(init_kernel, dim3(1024), dim3(256), 0, stream, h, acc);
    auto pack = [&](const float* W, unsigned short* dst, int K, int N, int Npad) {
        int total = K * Npad;
        hipLaunchKernelGGL(pack_w_kernel, dim3((total + 255) / 256), dim3(256), 0, stream,
                           W, dst, K, N, Npad);
    };
    pack(ode_Wr, Wr_pk, 256, 256, 256);
    pack(ode_Wz, Wz_pk, 256, 256, 256);
    pack(ode_Wh, Wh_pk, 256, 256, 256);
    pack(g_Uhr, Uhr_pk, 256, 256, 256);
    pack(g_Uhz, Uhz_pk, 256, 256, 256);
    pack(g_Uhn, Uhn_pk, 256, 256, 256);
    pack(p_W1, W1_pk, 256, 256, 256);
    pack(g_Wxr, Wxr_pk, 32, 256, 256);
    pack(g_Wxz, Wxz_pk, 32, 256, 256);
    pack(g_Wxn, Wxn_pk, 32, 256, 256);
    pack(p_W2, W2_pk, 256, 15, 16);

    for (int e = 0; e < 64; ++e) {
        hipLaunchKernelGGL(euler4_kernel, dim3(32), dim3(256), 0, stream,
                           h, Wr_pk, Wz_pk, Wh_pk, ode_br, ode_bz, ode_bh, dt);
        hipLaunchKernelGGL(event_kernel, dim3(16), dim3(256), 0, stream,
                           h, X, M, bidx, e,
                           Uhr_pk, Uhz_pk, Uhn_pk, Wxr_pk, Wxz_pk, Wxn_pk, W1_pk, W2_pk,
                           g_bxr, g_bxz, g_bxn, g_bhr, g_bhz, g_bhn, p_b1, p_b2, acc);
    }
    hipLaunchKernelGGL(final_kernel, dim3(1), dim3(32), 0, stream, acc, (float*)d_out);
}
